// NMSSoft_46574625357926
// MI455X (gfx1250) — compile-verified, matrix operand order checked by reading, not run on hardware
//
#include <hip/hip_runtime.h>
#include <hip/hip_bf16.h>
#include <math.h>

// Soft-NMS for N=8192 boxes, SIGMA=0.5, CONF_THRESH=0.0, class-aware.
//
// Strategy (MI455X):
//  1) nms_sort_kernel:   single-WG bitonic sort by confidence (desc), 64KB LDS.
//  2) nms_decay_kernel:  precompute full 8192x8192 bf16 decay matrix
//                        decay[i][j] = exp(-2*(iou * mask)^2), mask = (cls_i==cls_j)
//                        Class-equality tile computed EXACTLY via one-hot
//                        v_wmma_f32_16x16x32_f16 (K=96, 3 wmma per 16x16 tile);
//                        mask is applied BRANCHLESSLY as iou*c[r] (c[r] in {0,1}).
//                        128MB matrix fits in the 192MB global L2.
//  3) nms_copy_kernel:   out = boxes.
//  4) nms_serial_kernel: persistent 1024-thread WG; temps in registers
//                        (8/thread); per-iteration: argmax (shfl_xor + LDS)
//                        + 16B coalesced L2-resident decay-row load + 8 FMAs.
//
// Workspace requirement: 229376 + 8192*8192*2 bytes (~128.2 MB).

#define N_BOX 8192
#define SORT_TPB 1024
#define SER_TPB 1024
#define PER 8  // boxes per thread in serial kernel

typedef __attribute__((ext_vector_type(16))) _Float16 v16h;
typedef __attribute__((ext_vector_type(8)))  float    v8f;

__device__ __forceinline__ unsigned short f32_to_bf16(float f) {
  union { float f; unsigned u; } c; c.f = f;
  unsigned u = c.u + 0x7FFFu + ((c.u >> 16) & 1u);  // round-to-nearest-even
  return (unsigned short)(u >> 16);
}
__device__ __forceinline__ float bf16_to_f32(unsigned short h) {
  union { unsigned u; float f; } c; c.u = ((unsigned)h) << 16;
  return c.f;
}

// ---------------------------------------------------------------------------
// Kernel 1: bitonic sort (descending by confidence), emit sorted box data.
// ---------------------------------------------------------------------------
__global__ __launch_bounds__(SORT_TPB)
void nms_sort_kernel(const float* __restrict__ boxes,
                     float4* __restrict__ sbox,   // {x1,y1,x2,y2} sorted
                     float*  __restrict__ scls,
                     float*  __restrict__ sscore,
                     int*    __restrict__ sorder) {
  __shared__ float key[N_BOX];
  __shared__ int   idx[N_BOX];
  const int t = threadIdx.x;
  for (int p = 0; p < N_BOX / SORT_TPB; ++p) {
    int i = t + p * SORT_TPB;
    key[i] = boxes[i * 7 + 5];
    idx[i] = i;
  }
  for (unsigned k = 2; k <= N_BOX; k <<= 1) {
    for (unsigned j = k >> 1; j > 0; j >>= 1) {
      __syncthreads();
      for (int p = 0; p < N_BOX / SORT_TPB; ++p) {
        unsigned i = (unsigned)t + p * SORT_TPB;
        unsigned ixj = i ^ j;
        if (ixj > i) {
          bool up = ((i & k) == 0);  // "up" blocks sort descending
          float ki = key[i], kj = key[ixj];
          bool sw = up ? (ki < kj) : (ki > kj);
          if (sw) {
            key[i] = kj; key[ixj] = ki;
            int tmp = idx[i]; idx[i] = idx[ixj]; idx[ixj] = tmp;
          }
        }
      }
    }
  }
  __syncthreads();
  for (int p = 0; p < N_BOX / SORT_TPB; ++p) {
    int s = t + p * SORT_TPB;
    int o = idx[s];
    float cx = boxes[o * 7 + 1], cy = boxes[o * 7 + 2];
    float w  = boxes[o * 7 + 3], h  = boxes[o * 7 + 4];
    sbox[s]   = make_float4(cx - 0.5f * w, cy - 0.5f * h,
                            cx + 0.5f * w, cy + 0.5f * h);
    scls[s]   = boxes[o * 7 + 6];
    sscore[s] = key[s];
    sorder[s] = o;
  }
}

// ---------------------------------------------------------------------------
// Kernel 2: decay matrix precompute. One wave per 16x16 tile.
// Class mask via exact one-hot WMMA: E = onehot(cls_i) . onehot(cls_j)^T.
// Mask applied branchlessly: c[r] is exactly 0.0 or 1.0, and exp(0)=1 covers
// the class-mismatch case (reference: ious *= mask, then exp(-iou^2/sigma)).
// ---------------------------------------------------------------------------
__global__ __launch_bounds__(256)
void nms_decay_kernel(const float4* __restrict__ sbox,
                      const float*  __restrict__ scls,
                      unsigned short* __restrict__ decay) {
  const int lane   = threadIdx.x & 31;
  const int tile   = blockIdx.x * 8 + (threadIdx.x >> 5);
  const int ti     = tile >> 9;       // 512 tiles per dim
  const int tj     = tile & 511;
  const int i0     = ti * 16, j0 = tj * 16;
  const int l15    = lane & 15;
  const int hiHalf = lane >> 4;       // 0 or 1

  // A-matrix lane holds row M=lane&15; B-matrix lane holds col N=lane&15.
  const float clsA = scls[i0 + l15];
  const float clsB = scls[j0 + l15];

  v8f c = {};
  #pragma unroll
  for (int kk = 0; kk < 3; ++kk) {   // K = 96 (80 classes zero-padded)
    v16h a, b;
    #pragma unroll
    for (int hh = 0; hh < 16; ++hh) {
      // 16-bit A/B layout: half hh -> K = (hh<8 ? hh : hh+8) + 8*hiHalf
      int kidx = (hh < 8 ? hh : hh + 8) + hiHalf * 8 + kk * 32;
      a[hh] = (clsA == (float)kidx) ? (_Float16)1.0f : (_Float16)0.0f;
      b[hh] = (clsB == (float)kidx) ? (_Float16)1.0f : (_Float16)0.0f;
    }
    c = __builtin_amdgcn_wmma_f32_16x16x32_f16(
        /*neg_a=*/false, a, /*neg_b=*/false, b,
        /*c_mod=*/(short)0, c, /*reuse_a=*/false, /*reuse_b=*/false);
  }

  // IoU + decay for this lane's 8 (M,N) pairs; C/D layout: N=lane&15, M=r+8*hiHalf.
  const float4 bj = sbox[j0 + l15];
  const float areaB = (bj.z - bj.x) * (bj.w - bj.y);
  #pragma unroll
  for (int r = 0; r < 8; ++r) {
    const int mi = r + hiHalf * 8;
    const float4 bi = sbox[i0 + mi];
    const float areaA = (bi.z - bi.x) * (bi.w - bi.y);
    float dx = fmaxf(fminf(bi.z, bj.z) - fmaxf(bi.x, bj.x), 0.0f);
    float dy = fmaxf(fminf(bi.w, bj.w) - fmaxf(bi.y, bj.y), 0.0f);
    float inter = dx * dy;
    // masked IoU: c[r] == 1.0 iff classes match, else 0.0 (exact)
    float iou = (inter / (areaA + areaB - inter)) * c[r];
    // SIGMA=0.5 -> exp(-iou^2/sigma) = exp(-2*iou^2); exp(0)=1 for mismatch.
    float d = __expf(-2.0f * iou * iou);
    decay[(size_t)(i0 + mi) * N_BOX + (size_t)(j0 + l15)] = f32_to_bf16(d);
  }
}

// ---------------------------------------------------------------------------
// Kernel 3: copy boxes -> out (confidence column overwritten later).
// ---------------------------------------------------------------------------
__global__ void nms_copy_kernel(const float* __restrict__ in,
                                float* __restrict__ out, int n) {
  int i = blockIdx.x * blockDim.x + threadIdx.x;
  if (i < n) out[i] = in[i];
}

// ---------------------------------------------------------------------------
// Kernel 4: sequential soft-NMS sweep. Single 1024-thread workgroup.
// ---------------------------------------------------------------------------
__global__ __launch_bounds__(SER_TPB)
void nms_serial_kernel(const float* __restrict__ sscore,
                       const int*   __restrict__ sorder,
                       const unsigned short* __restrict__ decay,
                       float* __restrict__ out) {
  const int t    = threadIdx.x;
  const int lane = t & 31;
  const int wave = t >> 5;
  const int base = t * PER;

  float temp[PER], fin[PER];
  #pragma unroll
  for (int r = 0; r < PER; ++r) {
    temp[r] = sscore[base + r];
    fin[r]  = temp[r];
  }

  __shared__ float sVal[SER_TPB / 32];
  __shared__ int   sIdx[SER_TPB / 32];
  __shared__ float bVal;
  __shared__ int   bIdx;

  for (int iter = 0; iter < N_BOX; ++iter) {
    // --- local argmax over owned boxes (ties -> lowest index) ---
    float mv = -2.0f; int mi = 0x7FFFFFFF;
    #pragma unroll
    for (int r = 0; r < PER; ++r) {
      if (temp[r] > mv) { mv = temp[r]; mi = base + r; }
    }
    // --- wave32 reduction ---
    #pragma unroll
    for (int off = 16; off >= 1; off >>= 1) {
      float ov = __shfl_xor(mv, off, 32);
      int   oi = __shfl_xor(mi, off, 32);
      if (ov > mv || (ov == mv && oi < mi)) { mv = ov; mi = oi; }
    }
    if (lane == 0) { sVal[wave] = mv; sIdx[wave] = mi; }
    __syncthreads();
    // --- cross-wave reduction in wave 0 ---
    if (wave == 0) {
      mv = sVal[lane]; mi = sIdx[lane];
      #pragma unroll
      for (int off = 16; off >= 1; off >>= 1) {
        float ov = __shfl_xor(mv, off, 32);
        int   oi = __shfl_xor(mi, off, 32);
        if (ov > mv || (ov == mv && oi < mi)) { mv = ov; mi = oi; }
      }
      if (lane == 0) { bVal = mv; bIdx = mi; }
    }
    __syncthreads();
    const float selV = bVal;
    const int   sel  = bIdx;
    if (selV <= 0.0f) break;  // CONF_THRESH = 0.0; inactive forever after

    // Owner commits the selected box's score and removes it.
    if (sel >= base && sel < base + PER) {
      fin[sel - base]  = selV;
      temp[sel - base] = -1.0f;
    }

    // Decay all surviving boxes: one coalesced 16B load per thread (L2-resident).
    const uint4 v = *reinterpret_cast<const uint4*>(
        decay + (size_t)sel * N_BOX + (size_t)base);
    unsigned du[4] = {v.x, v.y, v.z, v.w};
    #pragma unroll
    for (int r = 0; r < PER; ++r) {
      unsigned short hb = (unsigned short)((du[r >> 1] >> ((r & 1) * 16)) & 0xFFFFu);
      if (temp[r] > -0.5f) temp[r] *= bf16_to_f32(hb);
    }
  }

  // Scatter final confidences back to original box order.
  #pragma unroll
  for (int r = 0; r < PER; ++r) {
    float fv = (temp[r] > -0.5f) ? temp[r] : fin[r];
    out[(size_t)sorder[base + r] * 7 + 5] = fv;
  }
}

// ---------------------------------------------------------------------------
extern "C" void kernel_launch(void* const* d_in, const int* in_sizes, int n_in,
                              void* d_out, int out_size, void* d_ws, size_t ws_size,
                              hipStream_t stream) {
  const float* boxes = (const float*)d_in[0];
  float* out = (float*)d_out;

  // Workspace layout (bytes):
  //   [0)        sbox   : N*16  (float4 x1,y1,x2,y2)
  //   [131072)   scls   : N*4
  //   [163840)   sscore : N*4
  //   [196608)   sorder : N*4
  //   [229376)   decay  : N*N*2 bf16 (128 MB)  -> total ~128.2 MB required.
  char* ws = (char*)d_ws;
  float4*         sbox   = (float4*)(ws);
  float*          scls   = (float*)(ws + (size_t)N_BOX * 16);
  float*          sscore = (float*)(ws + (size_t)N_BOX * 20);
  int*            sorder = (int*)  (ws + (size_t)N_BOX * 24);
  unsigned short* decay  = (unsigned short*)(ws + (size_t)N_BOX * 28);

  // 1) sort (single workgroup)
  nms_sort_kernel<<<1, SORT_TPB, 0, stream>>>(boxes, sbox, scls, sscore, sorder);

  // 2) decay matrix: 512x512 tiles of 16x16, one tile per wave, 8 waves/block
  nms_decay_kernel<<<(512 * 512) / 8, 256, 0, stream>>>(sbox, scls, decay);

  // 3) copy boxes to output
  const int ncopy = N_BOX * 7;
  nms_copy_kernel<<<(ncopy + 255) / 256, 256, 0, stream>>>(boxes, out, ncopy);

  // 4) sequential soft-NMS sweep (single workgroup)
  nms_serial_kernel<<<1, SER_TPB, 0, stream>>>(sscore, sorder, decay, out);
}
